// DepthConv_11785390260303
// MI455X (gfx1250) — matrix operand order repaired and checked by reading, not compiled, not run on hardware
//
#include <hip/hip_runtime.h>

typedef __attribute__((ext_vector_type(16))) _Float16 v16h;
typedef __attribute__((ext_vector_type(8)))  float    v8f;
typedef __attribute__((ext_vector_type(2)))  _Float16 h2;
typedef int v4i __attribute__((vector_size(16)));   // matches async-builtin param type

#define BB    8
#define CIN   128
#define COUT  128
#define HH    128
#define WWID  128
#define ALPHA 8.3f
#define LDSPITCH 136              // 128 halves + 8 pad -> 272B row stride, conflict-free b128 LDS reads
#define LDSBUF   (COUT * LDSPITCH)
#define MG    4                   // pixel groups per wave: 64 pixels x 128 Cout per wave

#if __has_builtin(__builtin_amdgcn_global_load_async_to_lds_b128)
#define USE_ASYNC 1
#else
#define USE_ASYNC 0
#endif

// ---- pack input: NCHW f32 -> NHWC f16 via LDS-tiled transpose (coalesced both ways) ----
#define TDIM 64
__global__ void __launch_bounds__(256)
pack_input_kernel(const float* __restrict__ in, _Float16* __restrict__ xws) {
    __shared__ float tile[TDIM][TDIM + 1];
    const int t  = threadIdx.x;
    const int w0 = blockIdx.x * TDIM;          // W/64 = 2
    const int c0 = blockIdx.y * TDIM;          // C/64 = 2
    const int b  = blockIdx.z / HH;            // B*H = 1024
    const int h  = blockIdx.z % HH;

    const int wl = t & 63;                      // read: w fastest (coalesced f32)
    const int cl = t >> 6;
#pragma unroll
    for (int r = 0; r < 16; ++r) {
        const int c = c0 + cl + r * 4;
        tile[cl + r * 4][wl] = in[(((size_t)b * CIN + c) * HH + h) * WWID + w0 + wl];
    }
    __syncthreads();

    const int cw = t & 63;                      // write: c fastest (coalesced f16)
    const int wr = t >> 6;
#pragma unroll
    for (int r = 0; r < 16; ++r) {
        const int w = w0 + wr + r * 4;
        xws[(((size_t)b * HH + h) * WWID + w) * CIN + c0 + cw] = (_Float16)tile[cw][wr + r * 4];
    }
}

// ---- pack weight: (Cout,Cin,3,3) f32 -> [tap][Cout][Cin] f16 ----
__global__ void pack_weight_kernel(const float* __restrict__ wt, _Float16* __restrict__ wws) {
    int i = blockIdx.x * blockDim.x + threadIdx.x;
    const int total = 9 * COUT * CIN;
    if (i >= total) return;
    int c = i % CIN; int r = i / CIN;
    int o = r % COUT; int tap = r / COUT;
    wws[i] = (_Float16)wt[((size_t)o * CIN + c) * 9 + tap];
}

// ---- stage one tap's 128x128 f16 weight matrix into LDS buffer `buf` ----
__device__ __forceinline__ void stage_tap(const _Float16* __restrict__ wws,
                                          _Float16* lds, int tap, int buf, int tid) {
    const uint4* src = (const uint4*)(wws + (size_t)tap * COUT * CIN);
    _Float16* dstbase = lds + buf * LDSBUF;
#pragma unroll
    for (int j = 0; j < 16; ++j) {
        int idx = j * 128 + tid;           // uint4 index within tap block (8 halves each)
        int o   = idx >> 4;                // 16 uint4 per 128-half row
        int c8  = (idx & 15) * 8;
        _Float16* dst = dstbase + o * LDSPITCH + c8;
#if USE_ASYNC
        __builtin_amdgcn_global_load_async_to_lds_b128(
            (__attribute__((address_space(1))) v4i*)(src + idx),
            (__attribute__((address_space(3))) v4i*)dst,
            0, 0);
#else
        *(uint4*)dst = src[idx];
#endif
    }
}

__device__ __forceinline__ void wait_stage_and_barrier() {
#if USE_ASYNC
#if __has_builtin(__builtin_amdgcn_s_wait_asynccnt)
    __builtin_amdgcn_s_wait_asynccnt(0);
#else
    asm volatile("s_wait_asynccnt 0" ::: "memory");
#endif
#endif
    __syncthreads();
}

// ---- main WMMA kernel: 1 wave -> 64 pixels x 128 Cout (B-frag reused 4x) ----
__global__ void __launch_bounds__(128)
depthconv_wmma_kernel(const _Float16* __restrict__ xws,
                      const _Float16* __restrict__ wws,
                      const float*    __restrict__ depth,
                      const float*    __restrict__ bias,
                      float*          __restrict__ out) {
    __shared__ __align__(16) _Float16 lds_w[2 * LDSBUF];   // double-buffered taps (~68 KB)

    const int lane = threadIdx.x;        // 0..31 (wave32)
    const int wvid = threadIdx.y;        // 0..3  (4 waves / block)
    const int tid  = wvid * 32 + lane;   // 0..127
    const int m    = lane & 15;
    const int hi   = lane >> 4;

    const int tile  = blockIdx.x * 4 + wvid;   // 64-pixel tile id (2048 total)
    const int pbase = tile * (MG * 16);
    const int b   = pbase / (HH * WWID);
    const int rem = pbase % (HH * WWID);
    const int h   = rem / WWID;
    const int w0  = rem % WWID;                // W=128 divisible by 64: tiles never straddle rows

    int wpx[MG];
    float dc[MG];
#pragma unroll
    for (int g = 0; g < MG; ++g) {
        wpx[g] = w0 + g * 16 + m;              // pixel group g (A-matrix rows M)
        dc[g]  = depth[(size_t)b * HH * WWID + h * WWID + wpx[g]];
    }

    // Accumulators start at bias: every element of acc[g][nt] maps to the same
    // Cout column o = nt*16 + m (D layout: lane -> N, VGPR -> M), so
    // acc_init = bias[o] makes the GEMM produce conv+bias directly.
    v8f acc[MG][8];
#pragma unroll
    for (int nt = 0; nt < 8; ++nt) {
        const float bv = bias[nt * 16 + m];
        const v8f binit = {bv, bv, bv, bv, bv, bv, bv, bv};
#pragma unroll
        for (int g = 0; g < MG; ++g) acc[g][nt] = binit;
    }

    stage_tap(wws, lds_w, 0, 0, tid);          // prefetch tap 0

    for (int tap = 0; tap < 9; ++tap) {
        wait_stage_and_barrier();              // current buffer ready across all waves
        if (tap < 8) stage_tap(wws, lds_w, tap + 1, (tap + 1) & 1, tid);  // async prefetch next

        const _Float16* wbuf = lds_w + (tap & 1) * LDSBUF;

        const int  kh  = tap / 3 - 1, kw = tap % 3 - 1;
        const int  hh2 = h + kh;
        const bool hin = (hh2 >= 0) & (hh2 < HH);
        const int  hcl = hh2 < 0 ? 0 : (hh2 >= HH ? HH - 1 : hh2);

        bool inb[MG];
        h2   sim2[MG];
        const _Float16* xrow[MG];
#pragma unroll
        for (int g = 0; g < MG; ++g) {
            const int ww2 = wpx[g] + kw;
            inb[g] = hin & (ww2 >= 0) & (ww2 < WWID);
            float dt = dc[g];                  // OOB: x=0 anyway, sim value irrelevant
            if (inb[g]) dt = depth[(size_t)b * HH * WWID + hh2 * WWID + ww2];
            const float    sim = __expf(-ALPHA * fabsf(dc[g] - dt));
            const _Float16 sh  = (_Float16)sim;
            sim2[g] = (h2){sh, sh};
            const int wcl = ww2 < 0 ? 0 : (ww2 >= WWID ? WWID - 1 : ww2);
            xrow[g] = xws + (((size_t)b * HH + hcl) * WWID + wcl) * CIN;
        }

#pragma unroll
        for (int ck = 0; ck < 4; ++ck) {
            // A fragments (16x32 f16, M=pixel): lanes 0-15 hold K {0..7,16..23},
            // lanes 16-31 hold K {8..15,24..31} of this 32-wide Cin chunk
            union { uint4 u[2]; h2 hp[8]; v16h v; } A[MG];
#pragma unroll
            for (int g = 0; g < MG; ++g) {
                const _Float16* ap = xrow[g] + ck * 32 + hi * 8;
                uint4 a0 = *(const uint4*)(ap);
                uint4 a1 = *(const uint4*)(ap + 16);
                const uint4 z = make_uint4(0u, 0u, 0u, 0u);
                A[g].u[0] = inb[g] ? a0 : z;
                A[g].u[1] = inb[g] ? a1 : z;
#pragma unroll
                for (int j = 0; j < 8; ++j) A[g].hp[j] *= sim2[g];   // v_pk_mul_f16
            }

#pragma unroll
            for (int nt = 0; nt < 8; ++nt) {
                // B fragment (32x16 f16, N=Cout): lane holds column o, K {0..7,16..23} (lanes 0-15) / {8..15,24..31} (lanes 16-31)
                const int o = nt * 16 + m;
                union { uint4 u[2]; v16h v; } Bf;
                const _Float16* wp = wbuf + o * LDSPITCH + ck * 32 + hi * 8;
                Bf.u[0] = *(const uint4*)(wp);
                Bf.u[1] = *(const uint4*)(wp + 16);
#pragma unroll
                for (int g = 0; g < MG; ++g) {
                    acc[g][nt] = __builtin_amdgcn_wmma_f32_16x16x32_f16(
                        false, A[g].v, false, Bf.v, (short)0, acc[g][nt], false, false);
                }
            }
        }
    }

    // epilogue: D layout = lane -> N (Cout col), VGPR r -> M (pixel r + hi*8)
#pragma unroll
    for (int nt = 0; nt < 8; ++nt) {
        const int o = nt * 16 + m;
#pragma unroll
        for (int g = 0; g < MG; ++g) {
            float* op = out + (((size_t)b * COUT + o) * HH + h) * WWID + w0 + g * 16 + hi * 8;
            union { v8f v; float4 f[2]; } U;
            U.v = acc[g][nt];
            *(float4*)(op)     = U.f[0];
            *(float4*)(op + 4) = U.f[1];
        }
    }
}

extern "C" void kernel_launch(void* const* d_in, const int* in_sizes, int n_in,
                              void* d_out, int out_size, void* d_ws, size_t ws_size,
                              hipStream_t stream) {
    const float* input  = (const float*)d_in[0];   // (B,Cin,H,W) f32
    const float* depth  = (const float*)d_in[1];   // (B,1,H,W)  f32
    const float* weight = (const float*)d_in[2];   // (Cout,Cin,3,3) f32
    const float* bias   = (const float*)d_in[3];   // (Cout,) f32
    float* out = (float*)d_out;                    // (B,Cout,H,W) f32

    _Float16* xws = (_Float16*)d_ws;                         // 32 MB: NHWC f16 input
    _Float16* wws = xws + (size_t)BB * CIN * HH * WWID;      // 288 KB: packed f16 weights

    {
        dim3 grd(WWID / TDIM, CIN / TDIM, BB * HH);          // (2, 2, 1024)
        pack_input_kernel<<<grd, 256, 0, stream>>>(input, xws);
    }
    {
        int total = 9 * COUT * CIN;
        pack_weight_kernel<<<(total + 255) / 256, 256, 0, stream>>>(weight, wws);
    }
    {
        dim3 blk(32, 4);                        // 4 wave32 per block
        int tiles = BB * HH * WWID / (MG * 16); // 2048 64-pixel tiles
        depthconv_wmma_kernel<<<tiles / 4, blk, 0, stream>>>(xws, wws, depth, bias, out);
    }
}
